// PowModel_31507880084188
// MI455X (gfx1250) — 8-bit matrix operand order repaired and checked by reading, compiled, not run on hardware
//
#include <hip/hip_runtime.h>
#include <hip/hip_fp16.h>


typedef int   v16i __attribute__((ext_vector_type(16)));
typedef float v8f  __attribute__((ext_vector_type(8)));

#define IN_SIZE        256
#define ROUNDS         64
#define ROWS_PER_BLOCK 16
#define THREADS        128

#define FP8_ONE  0x38u
#define FP8_NEG1 0xB8u

__global__ void prep_w_kernel(const float* __restrict__ W,
                              unsigned int* __restrict__ packed) {
  int t    = blockIdx.x * blockDim.x + threadIdx.x;
  int v    = t & 15;
  int lane = (t >> 4) & 31;
  int k    = (t >> 9) & 1;
  int n    = t >> 10;
  int Ncol = n * 16 + (lane & 15);
  unsigned int d = 0;
  #pragma unroll
  for (int b = 0; b < 4; ++b) {
    int Klocal = 32 * (v >> 2) + 16 * (lane >> 4) + ((v & 3) << 2) + b;
    int K      = k * 128 + Klocal;
    float w    = W[K * IN_SIZE + Ncol];
    unsigned int e = (w > 0.5f) ? FP8_ONE : ((w < -0.5f) ? FP8_NEG1 : 0u);
    d |= e << (8 * b);
  }
  packed[t] = d;
}

__global__ __launch_bounds__(THREADS)
void pow_rounds_kernel(const float* __restrict__ x,
                       const float* __restrict__ noise,
                       const float* __restrict__ bias,
                       const unsigned int* __restrict__ packedW,
                       float* __restrict__ out) {
  __shared__ __align__(16) unsigned char act[2][ROWS_PER_BLOCK][IN_SIZE];

  const int tid     = threadIdx.x;
  const int lane    = tid & 31;
  const int wave    = tid >> 5;
  const int rowBase = blockIdx.x * ROWS_PER_BLOCK;
  const int rowHalf = (lane >> 4) * 8;

  v16i B[4][2];
  #pragma unroll
  for (int n = 0; n < 4; ++n) {
    #pragma unroll
    for (int k = 0; k < 2; ++k) {
      int frag = (4 * wave + n) * 2 + k;
      const uint4* p = (const uint4*)(packedW + (frag * 32 + lane) * 16);
      #pragma unroll
      for (int q = 0; q < 4; ++q) {
        uint4 u = p[q];
        B[n][k][4 * q + 0] = (int)u.x;
        B[n][k][4 * q + 1] = (int)u.y;
        B[n][k][4 * q + 2] = (int)u.z;
        B[n][k][4 * q + 3] = (int)u.w;
      }
    }
  }

  float nb[4][8];
  #pragma unroll
  for (int n = 0; n < 4; ++n) {
    int col = (4 * wave + n) * 16 + (lane & 15);
    __half hb = __float2half(bias[col]);
    #pragma unroll
    for (int r = 0; r < 8; ++r) {
      int row = rowBase + r + rowHalf;
      __half hn = __float2half(noise[row * IN_SIZE + col]);
      nb[n][r] = -0.5f * __half2float(__hadd(hb, hn));
    }
  }

  #pragma unroll
  for (int i = 0; i < 8; ++i) {
    int dw = i * THREADS + tid;
    int r  = dw >> 6;
    int c4 = (dw & 63) * 4;
    const float* xp = x + (rowBase + r) * IN_SIZE + c4;
    unsigned int d = 0;
    d |= (xp[0] > 0.5f ? FP8_ONE : 0u);
    d |= (xp[1] > 0.5f ? FP8_ONE : 0u) << 8;
    d |= (xp[2] > 0.5f ? FP8_ONE : 0u) << 16;
    d |= (xp[3] > 0.5f ? FP8_ONE : 0u) << 24;
    *(unsigned int*)&act[0][r][c4] = d;
  }
  __syncthreads();

  const int arow  = lane & 15;
  const int aoff8 = (lane >> 4) * 8;

  int cur = 0;
  #pragma unroll 2
  for (int rd = 0; rd < ROUNDS; ++rd) {
    v16i A[2];
    #pragma unroll
    for (int k = 0; k < 2; ++k) {
      #pragma unroll
      for (int h = 0; h < 2; ++h) {
        #pragma unroll
        for (int j = 0; j < 4; ++j) {
          unsigned long long q = *(const unsigned long long*)
              &act[cur][arow][k * 128 + h * 64 + (j >> 1) * 32 + (j & 1) * 8 + 2 * aoff8];
          A[k][2 * (4 * h + j) + 0] = (int)(unsigned int)q;
          A[k][2 * (4 * h + j) + 1] = (int)(unsigned int)(q >> 32);
        }
      }
    }

    const int nxt = cur ^ 1;
    #pragma unroll
    for (int n = 0; n < 4; ++n) {
      v8f acc = {0.f, 0.f, 0.f, 0.f, 0.f, 0.f, 0.f, 0.f};
      acc = __builtin_amdgcn_wmma_f32_16x16x128_fp8_fp8(A[0], B[n][0],
                                                        (short)0, acc, false, false);
      acc = __builtin_amdgcn_wmma_f32_16x16x128_fp8_fp8(A[1], B[n][1],
                                                        (short)0, acc, false, false);
      int col = (4 * wave + n) * 16 + (lane & 15);
      if (rd < ROUNDS - 1) {
        #pragma unroll
        for (int r = 0; r < 8; ++r) {
          act[nxt][r + rowHalf][col] = (acc[r] > nb[n][r]) ? (unsigned char)FP8_ONE
                                                           : (unsigned char)0;
        }
      } else {
        #pragma unroll
        for (int r = 0; r < 8; ++r) {
          out[(rowBase + r + rowHalf) * IN_SIZE + col] =
              (acc[r] > nb[n][r]) ? 1.0f : 0.0f;
        }
      }
    }
    __syncthreads();
    cur = nxt;
  }
}

extern "C" void kernel_launch(void* const* d_in, const int* in_sizes, int n_in,
                              void* d_out, int out_size, void* d_ws, size_t ws_size,
                              hipStream_t stream) {
  const float* x     = (const float*)d_in[0];
  const float* noise = (const float*)d_in[1];
  const float* W     = (const float*)d_in[2];
  const float* bias  = (const float*)d_in[3];

  unsigned int* packedW = (unsigned int*)d_ws;

  prep_w_kernel<<<64, 256, 0, stream>>>(W, packedW);

  int batch = in_sizes[0] / IN_SIZE;
  pow_rounds_kernel<<<batch / ROWS_PER_BLOCK, THREADS, 0, stream>>>(
      x, noise, bias, packedW, (float*)d_out);
}
